// LSTM_SPLID_69157563400850
// MI455X (gfx1250) — matrix operand order repaired and checked by reading, not compiled, not run on hardware
//
#include <hip/hip_runtime.h>

typedef __attribute__((ext_vector_type(16))) _Float16 v16h;
typedef __attribute__((ext_vector_type(8)))  _Float16 v8h;
typedef __attribute__((ext_vector_type(8)))  float    v8f;

static constexpr int B_TOT = 32768;
static constexpr int T_SEQ = 25;
static constexpr int IN_D  = 16;
static constexpr int H_D   = 64;
static constexpr int G4    = 256;   // 4*H
static constexpr int KFC   = 1600;  // T*H

static constexpr size_t WS_FLAT_BYTES = (size_t)B_TOT * KFC * sizeof(_Float16); // 104857600
static constexpr int    FCW_ELEMS     = 128 * KFC;                              // per head

// ---------------------------------------------------------------------------
// WMMA helper:  D = A(16x32 f16) * B(32x16 f16) + C(16x16 f32)
// ---------------------------------------------------------------------------
__device__ __forceinline__ v8f wmma_f16(v16h a, v16h b, v8f c) {
  return __builtin_amdgcn_wmma_f32_16x16x32_f16(
      /*neg_a=*/false, a, /*neg_b=*/false, b,
      /*c_mod=*/(short)0, c, /*reuse_a=*/false, /*reuse_b=*/false);
}

// ---------------------------------------------------------------------------
// A-fragment (16x32, f16) gather from a row-major f16 matrix (LDS or global).
// ISA 7.12.2: lanes 0-15 -> K 0..7 / 16..23, lanes 16-31 -> K 8..15 / 24..31.
// Rows must be 16B aligned (lda multiple of 8 halves).
// ---------------------------------------------------------------------------
__device__ __forceinline__ v16h loadA16(const _Float16* base, int lda, int lane, int k0) {
  int row = lane & 15;
  int kb  = k0 + ((lane & 16) ? 8 : 0);
  const _Float16* p = base + row * lda + kb;
  v8h lo = *(const v8h*)p;
  v8h hi = *(const v8h*)(p + 16);
  v16h a;
#pragma unroll
  for (int j = 0; j < 8; ++j) { a[j] = lo[j]; a[8 + j] = hi[j]; }
  return a;
}

// ---------------------------------------------------------------------------
// B-fragment (32x16) from row-major *f16* weights W[N][K]: lane holds column n,
// K 0..7 / 16..23 (lanes 0-15) or 8..15 / 24..31 (lanes 16-31) -> two 16B vector loads.
// ---------------------------------------------------------------------------
__device__ __forceinline__ v16h loadB16(const _Float16* __restrict__ W, int K,
                                        int n, int k0, int lane) {
  int kb = k0 + ((lane & 16) ? 8 : 0);
  const _Float16* p = W + (size_t)n * K + kb;
  v8h lo = *(const v8h*)p;
  v8h hi = *(const v8h*)(p + 16);
  v16h b;
#pragma unroll
  for (int j = 0; j < 8; ++j) { b[j] = lo[j]; b[8 + j] = hi[j]; }
  return b;
}

// B-fragment from row-major f32 weights W[N][K] (small, one-time, loop-hoisted).
__device__ __forceinline__ v16h loadBf32(const float* __restrict__ W, int K,
                                         int n, int k0, int lane, int kmax) {
  int kb = k0 + ((lane & 16) ? 8 : 0);
  v16h b;
#pragma unroll
  for (int j = 0; j < 16; ++j) {
    int k = kb + (j < 8 ? j : j + 8);
    float v = (k < kmax) ? W[n * K + k] : 0.f;
    b[j] = (_Float16)v;
  }
  return b;
}

// B-fragment for the tiny output layers W[nout][128] with column zero-fill.
__device__ __forceinline__ v16h loadBsmall(const float* __restrict__ W, int K,
                                           int ncol, int nmax, int k0, int lane) {
  int kb = k0 + ((lane & 16) ? 8 : 0);
  v16h b;
#pragma unroll
  for (int j = 0; j < 16; ++j) {
    float v = (ncol < nmax) ? W[ncol * K + kb + (j < 8 ? j : j + 8)] : 0.f;
    b[j] = (_Float16)v;
  }
  return b;
}

__device__ __forceinline__ float sigmoidf(float v) {
  return 1.f / (1.f + __expf(-v));
}

// Branch-free tanh: 1 - 2/(1 + e^{2x}); saturates correctly via exp over/underflow.
__device__ __forceinline__ float fast_tanhf(float v) {
  return 1.f - 2.f / (1.f + __expf(2.f * v));
}

// ===========================================================================
// Prep: f32 -> f16 weight conversion (FC weights, done once, 1.2 MB total)
// ===========================================================================
__global__ __launch_bounds__(256) void f32_to_f16_kernel(
    const float* __restrict__ src, _Float16* __restrict__ dst, int n) {
  int i = blockIdx.x * 256 + threadIdx.x;
  if (i < n) dst[i] = (_Float16)src[i];
}

// ===========================================================================
// Kernel 1: fused 2-layer LSTM for a 16-sample batch tile.
// Block: 128 threads = 4 waves; wave w computes gate w (i,f,g,o):
// gate columns [64w, 64w+64) as 4 WMMA N-tiles.
// Gate staging buffer is TRANSPOSED: sh_g[n][m] so each lane's 8 accumulator
// values (m = hi16*8..+7, fixed n) store as one packed ds_store_b128.
// ===========================================================================
__global__ __launch_bounds__(128) void lstm2_kernel(
    const float* __restrict__ x,
    const float* __restrict__ w_ih0, const float* __restrict__ w_hh0,
    const float* __restrict__ b_ih0, const float* __restrict__ b_hh0,
    const float* __restrict__ w_ih1, const float* __restrict__ w_hh1,
    const float* __restrict__ b_ih1, const float* __restrict__ b_hh1,
    _Float16* __restrict__ flat_out)   // [B][T*H] f16
{
  __shared__ _Float16 sh_h[16][H_D];            // current hidden state (A source)
  __shared__ _Float16 sh_g[G4][16];             // gate pre-activations, transposed
  __shared__ _Float16 sh_h0[T_SEQ][16][H_D];    // layer-0 hidden states, all t

  const int tid  = threadIdx.x;
  const int lane = tid & 31;
  const int wave = tid >> 5;
  const int b0   = blockIdx.x * 16;
  const int n0w  = wave * 64;
  const int ncol = lane & 15;
  const int hi16 = (lane & 16) ? 1 : 0;

  // elementwise ownership: thread owns (m,j) = (idx>>6, idx&63), idx = tid + 128*i
  float creg[8];
  float bsum[4][8];
#pragma unroll
  for (int i = 0; i < 8; ++i) {
    int j = (tid + 128 * i) & 63;
    creg[i] = 0.f;
#pragma unroll
    for (int g = 0; g < 4; ++g) bsum[g][i] = b_ih0[g * 64 + j] + b_hh0[g * 64 + j];
  }

  for (int idx = tid; idx < 16 * H_D; idx += 128) ((_Float16*)sh_h)[idx] = (_Float16)0.f;
  __syncthreads();

  // =========================== layer 0 ===========================
  {
    v16h Bih[4], Bhh[4][2];
#pragma unroll
    for (int nt = 0; nt < 4; ++nt) {
      int n = n0w + nt * 16 + ncol;
      Bih[nt]    = loadBf32(w_ih0, IN_D, n, 0, lane, IN_D);   // K=16 padded to 32
      Bhh[nt][0] = loadBf32(w_hh0, H_D,  n, 0, lane, H_D);
      Bhh[nt][1] = loadBf32(w_hh0, H_D,  n, 32, lane, H_D);
    }

    for (int t = 0; t < T_SEQ; ++t) {
      v16h ax;   // x_t fragment, K=16 real, padded
      {
        int row = lane & 15;
        const float* xr = x + ((size_t)(b0 + row) * T_SEQ + t) * IN_D + (hi16 ? 8 : 0);
#pragma unroll
        for (int j = 0; j < 8; ++j) { ax[j] = (_Float16)xr[j]; ax[8 + j] = (_Float16)0.f; }
      }
      v16h ah0 = loadA16(&sh_h[0][0], H_D, lane, 0);
      v16h ah1 = loadA16(&sh_h[0][0], H_D, lane, 32);

      v8f acc[4];
#pragma unroll
      for (int nt = 0; nt < 4; ++nt) {
        v8f a = {};
        a = wmma_f16(ax,  Bih[nt],    a);
        a = wmma_f16(ah0, Bhh[nt][0], a);
        a = wmma_f16(ah1, Bhh[nt][1], a);
        acc[nt] = a;
      }
#pragma unroll
      for (int nt = 0; nt < 4; ++nt) {
        int n = n0w + nt * 16 + ncol;
        v8h hv;
#pragma unroll
        for (int r = 0; r < 8; ++r) hv[r] = (_Float16)acc[nt][r];
        *(v8h*)&sh_g[n][hi16 * 8] = hv;          // one ds_store_b128
      }
      __syncthreads();   // sh_g complete; also fences sh_h reads above

#pragma unroll
      for (int i = 0; i < 8; ++i) {
        int idx = tid + 128 * i;
        int m = idx >> 6, j = idx & 63;
        float ig = sigmoidf  ((float)sh_g[j][m]       + bsum[0][i]);
        float fg = sigmoidf  ((float)sh_g[64 + j][m]  + bsum[1][i]);
        float gg = fast_tanhf((float)sh_g[128 + j][m] + bsum[2][i]);
        float og = sigmoidf  ((float)sh_g[192 + j][m] + bsum[3][i]);
        float c  = fg * creg[i] + ig * gg;
        creg[i]  = c;
        _Float16 h = (_Float16)(og * fast_tanhf(c));
        sh_h[m][j]     = h;
        sh_h0[t][m][j] = h;
      }
      __syncthreads();
    }
  }

  // =========================== layer 1 ===========================
  {
#pragma unroll
    for (int i = 0; i < 8; ++i) {
      int j = (tid + 128 * i) & 63;
      creg[i] = 0.f;
#pragma unroll
      for (int g = 0; g < 4; ++g) bsum[g][i] = b_ih1[g * 64 + j] + b_hh1[g * 64 + j];
    }
    for (int idx = tid; idx < 16 * H_D; idx += 128) ((_Float16*)sh_h)[idx] = (_Float16)0.f;
    __syncthreads();

    v16h Bih1[4][2], Bhh1[4][2];
#pragma unroll
    for (int nt = 0; nt < 4; ++nt) {
      int n = n0w + nt * 16 + ncol;
      Bih1[nt][0] = loadBf32(w_ih1, H_D, n, 0, lane, H_D);
      Bih1[nt][1] = loadBf32(w_ih1, H_D, n, 32, lane, H_D);
      Bhh1[nt][0] = loadBf32(w_hh1, H_D, n, 0, lane, H_D);
      Bhh1[nt][1] = loadBf32(w_hh1, H_D, n, 32, lane, H_D);
    }

    for (int t = 0; t < T_SEQ; ++t) {
      v16h ax0 = loadA16(&sh_h0[t][0][0], H_D, lane, 0);
      v16h ax1 = loadA16(&sh_h0[t][0][0], H_D, lane, 32);
      v16h ah0 = loadA16(&sh_h[0][0],     H_D, lane, 0);
      v16h ah1 = loadA16(&sh_h[0][0],     H_D, lane, 32);

      v8f acc[4];
#pragma unroll
      for (int nt = 0; nt < 4; ++nt) {
        v8f a = {};
        a = wmma_f16(ax0, Bih1[nt][0], a);
        a = wmma_f16(ax1, Bih1[nt][1], a);
        a = wmma_f16(ah0, Bhh1[nt][0], a);
        a = wmma_f16(ah1, Bhh1[nt][1], a);
        acc[nt] = a;
      }
#pragma unroll
      for (int nt = 0; nt < 4; ++nt) {
        int n = n0w + nt * 16 + ncol;
        v8h hv;
#pragma unroll
        for (int r = 0; r < 8; ++r) hv[r] = (_Float16)acc[nt][r];
        *(v8h*)&sh_g[n][hi16 * 8] = hv;
      }
      __syncthreads();

#pragma unroll
      for (int i = 0; i < 8; ++i) {
        int idx = tid + 128 * i;
        int m = idx >> 6, j = idx & 63;
        float ig = sigmoidf  ((float)sh_g[j][m]       + bsum[0][i]);
        float fg = sigmoidf  ((float)sh_g[64 + j][m]  + bsum[1][i]);
        float gg = fast_tanhf((float)sh_g[128 + j][m] + bsum[2][i]);
        float og = sigmoidf  ((float)sh_g[192 + j][m] + bsum[3][i]);
        float c  = fg * creg[i] + ig * gg;
        creg[i]  = c;
        _Float16 h = (_Float16)(og * fast_tanhf(c));
        sh_h[m][j] = h;
        flat_out[(size_t)(b0 + m) * KFC + t * H_D + j] = h;   // [B, T*H]
      }
      __syncthreads();
    }
  }
}

// ===========================================================================
// Kernel 2: FC head GEMM (M=16 tile, N=128, K=1600) + ReLU + tiny projection.
// Grid: (B/16, 3 heads), 128 threads = 4 waves, 2 N-tiles each, 100 WMMAs/wave.
// ===========================================================================
struct FCArgs {
  const _Float16* A;              // [B][1600] f16 (workspace)
  const _Float16* fcwh;           // [3][128][1600] f16 (workspace)
  const float* fcb[3];            // [128]
  const float* lw[3];             // [nout][128]
  const float* lb[3];             // [nout]
  float*       out[3];            // [B][nout]
  int          nout[3];
};

__global__ __launch_bounds__(128) void fc_heads_kernel(FCArgs args) {
  __shared__ _Float16 shx[16][128];

  const int head = blockIdx.y;
  const int b0   = blockIdx.x * 16;
  const int tid  = threadIdx.x;
  const int wave = tid >> 5;
  const int lane = tid & 31;
  const int ncol = lane & 15;
  const int hi16 = (lane & 16) ? 1 : 0;
  const int n0   = wave * 32;

  const _Float16* __restrict__ W = args.fcwh + (size_t)head * FCW_ELEMS;
  const _Float16* Abase = args.A + (size_t)b0 * KFC;

  v8f acc0 = {}, acc1 = {};
  for (int k0 = 0; k0 < KFC; k0 += 32) {
    v16h a  = loadA16(Abase, KFC, lane, k0);
    v16h bA = loadB16(W, KFC, n0 + ncol,      k0, lane);
    v16h bB = loadB16(W, KFC, n0 + 16 + ncol, k0, lane);
    acc0 = wmma_f16(a, bA, acc0);
    acc1 = wmma_f16(a, bB, acc1);
  }

  // bias + ReLU -> LDS f16 tile [16][128]
  const float* fb = args.fcb[head];
  {
    int nA = n0 + ncol, nB = n0 + 16 + ncol;
    float biasA = fb[nA], biasB = fb[nB];
#pragma unroll
    for (int r = 0; r < 8; ++r) {
      int m = r + hi16 * 8;
      shx[m][nA] = (_Float16)fmaxf(acc0[r] + biasA, 0.f);
      shx[m][nB] = (_Float16)fmaxf(acc1[r] + biasB, 0.f);
    }
  }
  __syncthreads();

  // tiny projection 128 -> nout by wave 0 (wave-uniform branch; EXEC all-ones)
  if (wave == 0) {
    const int nout = args.nout[head];
    v8f o = {};
#pragma unroll
    for (int kc = 0; kc < 4; ++kc) {
      v16h a  = loadA16(&shx[0][0], 128, lane, kc * 32);
      v16h bb = loadBsmall(args.lw[head], 128, ncol, nout, kc * 32, lane);
      o = wmma_f16(a, bb, o);
    }
    if (ncol < nout) {
      float bias = args.lb[head][ncol];
      float* outp = args.out[head];
#pragma unroll
      for (int r = 0; r < 8; ++r) {
        int m = r + hi16 * 8;
        outp[(size_t)(b0 + m) * nout + ncol] = o[r] + bias;
      }
    }
  }
}

// ===========================================================================
extern "C" void kernel_launch(void* const* d_in, const int* in_sizes, int n_in,
                              void* d_out, int out_size, void* d_ws, size_t ws_size,
                              hipStream_t stream) {
  const float* x     = (const float*)d_in[0];
  const float* w_ih0 = (const float*)d_in[1];
  const float* w_hh0 = (const float*)d_in[2];
  const float* b_ih0 = (const float*)d_in[3];
  const float* b_hh0 = (const float*)d_in[4];
  const float* w_ih1 = (const float*)d_in[5];
  const float* w_hh1 = (const float*)d_in[6];
  const float* b_ih1 = (const float*)d_in[7];
  const float* b_hh1 = (const float*)d_in[8];
  const float* fc1_w = (const float*)d_in[9];
  const float* fc1_b = (const float*)d_in[10];
  const float* fc2_w = (const float*)d_in[11];
  const float* fc2_b = (const float*)d_in[12];
  const float* fc3_w = (const float*)d_in[13];
  const float* fc3_b = (const float*)d_in[14];
  const float* l1_w  = (const float*)d_in[15];
  const float* l1_b  = (const float*)d_in[16];
  const float* l2_w  = (const float*)d_in[17];
  const float* l2_b  = (const float*)d_in[18];
  const float* l3_w  = (const float*)d_in[19];
  const float* l3_b  = (const float*)d_in[20];

  _Float16* flat = (_Float16*)d_ws;                              // 100 MB
  _Float16* fcwh = (_Float16*)((char*)d_ws + WS_FLAT_BYTES);     // 1.2 MB
  float*    out  = (float*)d_out;

  // FC weight conversion (independent of the LSTM kernel; same stream)
  {
    int n = FCW_ELEMS, blocks = (n + 255) / 256;
    f32_to_f16_kernel<<<blocks, 256, 0, stream>>>(fc1_w, fcwh + 0 * (size_t)FCW_ELEMS, n);
    f32_to_f16_kernel<<<blocks, 256, 0, stream>>>(fc2_w, fcwh + 1 * (size_t)FCW_ELEMS, n);
    f32_to_f16_kernel<<<blocks, 256, 0, stream>>>(fc3_w, fcwh + 2 * (size_t)FCW_ELEMS, n);
  }

  lstm2_kernel<<<dim3(B_TOT / 16), 128, 0, stream>>>(
      x, w_ih0, w_hh0, b_ih0, b_hh0, w_ih1, w_hh1, b_ih1, b_hh1, flat);

  FCArgs fa;
  fa.A = flat;
  fa.fcwh = fcwh;
  fa.fcb[0] = fc1_b; fa.fcb[1] = fc2_b; fa.fcb[2] = fc3_b;
  fa.lw[0]  = l1_w;  fa.lw[1]  = l2_w;  fa.lw[2]  = l3_w;
  fa.lb[0]  = l1_b;  fa.lb[1]  = l2_b;  fa.lb[2]  = l3_b;
  fa.nout[0] = 5; fa.nout[1] = 4; fa.nout[2] = 8;
  fa.out[0] = out;
  fa.out[1] = out + (size_t)B_TOT * 5;
  fa.out[2] = out + (size_t)B_TOT * 9;

  fc_heads_kernel<<<dim3(B_TOT / 16, 3), 128, 0, stream>>>(fa);
}
